// TaggingQuantizer_62594853372134
// MI455X (gfx1250) — compile-verified, matrix operand order checked by reading, not run on hardware
//
#include <hip/hip_runtime.h>
#include <hip/hip_bf16.h>

#define N_E   1024
#define E_DIM 256
#define N_TOK 65536
#define BETA  0.25f

typedef __attribute__((ext_vector_type(16))) __bf16 v16bf;
typedef __attribute__((ext_vector_type(8)))  __bf16 v8bf;
typedef __attribute__((ext_vector_type(8)))  float  v8f;
typedef __attribute__((ext_vector_type(4)))  float  f32x4;
typedef __attribute__((ext_vector_type(4)))  unsigned int u32x4;

__device__ __forceinline__ __bf16 f32_to_bf16_rne(float f) {
    union { float f; unsigned u; } v; v.f = f;
    unsigned r = v.u + 0x7FFFu + ((v.u >> 16) & 1u);
    unsigned short h = (unsigned short)(r >> 16);
    return __builtin_bit_cast(__bf16, h);
}

// ---------------------------------------------------------------- init mask
__global__ void __launch_bounds__(1024) k_init(int* __restrict__ mask) {
    int t = threadIdx.x;
    if (t < N_E) mask[t] = 0;
}

// ------------------------------------------- normalize codebook rows -> bf16
__global__ void __launch_bounds__(256) k_normalize_W(const float* __restrict__ W,
                                                     __bf16* __restrict__ Wn) {
    int e = blockIdx.x;
    int t = threadIdx.x;
    float w = W[(size_t)e * E_DIM + t];
    float s = w * w;
    #pragma unroll
    for (int off = 16; off; off >>= 1) s += __shfl_xor(s, off, 32);
    __shared__ float sm[8];
    int wave = t >> 5, lane = t & 31;
    if (lane == 0) sm[wave] = s;
    __syncthreads();
    float tot = 0.f;
    #pragma unroll
    for (int i = 0; i < 8; ++i) tot += sm[i];
    float rn = rsqrtf(tot);
    Wn[(size_t)e * E_DIM + t] = f32_to_bf16_rne(w * rn);
}

// --------------------- WMMA scoring: per-token argmax over 1024 code vectors
// One wave owns a 32-row tile of z (two A fragment sets resident in VGPRs,
// bf16), streams all 64 16-column tiles of Wn once, and each B fetch feeds
// TWO wmma accumulators (1 load : 1 wmma issue ratio).
__global__ void __launch_bounds__(256) k_score_argmax(const float*  __restrict__ z,
                                                      const __bf16* __restrict__ Wn,
                                                      int* __restrict__ min_idx) {
    const int lane = threadIdx.x & 31;
    const int wave = threadIdx.x >> 5;
    const int gw   = blockIdx.x * 8 + wave;   // global wave -> 32-row tile
    const int m0   = gw * 32;
    const int rsel = lane & 15;               // A: row within tile; B: col within tile
    const int kb   = (lane >> 4) * 8;         // per-lane K sub-offset (ISA 16-bit layout)

    // Load 32 z-rows as bf16 A fragments: 2 row-halves x 8 k-steps of 32.
    v16bf a0[8], a1[8];
    const float* zrow0 = z + (size_t)(m0 + rsel) * E_DIM;
    const float* zrow1 = zrow0 + (size_t)16 * E_DIM;
    #pragma unroll
    for (int ks = 0; ks < 8; ++ks) {
        const int k0 = ks * 32 + kb;
        float t0[16], t1[16];
        *(f32x4*)(&t0[0])  = *(const f32x4*)(zrow0 + k0);
        *(f32x4*)(&t0[4])  = *(const f32x4*)(zrow0 + k0 + 4);
        *(f32x4*)(&t0[8])  = *(const f32x4*)(zrow0 + k0 + 16);
        *(f32x4*)(&t0[12]) = *(const f32x4*)(zrow0 + k0 + 20);
        *(f32x4*)(&t1[0])  = *(const f32x4*)(zrow1 + k0);
        *(f32x4*)(&t1[4])  = *(const f32x4*)(zrow1 + k0 + 4);
        *(f32x4*)(&t1[8])  = *(const f32x4*)(zrow1 + k0 + 16);
        *(f32x4*)(&t1[12]) = *(const f32x4*)(zrow1 + k0 + 20);
        #pragma unroll
        for (int j = 0; j < 16; ++j) a0[ks][j] = f32_to_bf16_rne(t0[j]);
        #pragma unroll
        for (int j = 0; j < 16; ++j) a1[ks][j] = f32_to_bf16_rne(t1[j]);
    }

    float bestv0[8], bestv1[8];
    int   besti0[8], besti1[8];
    #pragma unroll
    for (int r = 0; r < 8; ++r) {
        bestv0[r] = -3.4e38f; besti0[r] = 0;
        bestv1[r] = -3.4e38f; besti1[r] = 0;
    }

    for (int n = 0; n < N_E / 16; ++n) {
        const __bf16* wrow = Wn + (size_t)(n * 16 + rsel) * E_DIM;
        v8f c0 = {}, c1 = {};
        #pragma unroll
        for (int ks = 0; ks < 8; ++ks) {
            const int k0 = ks * 32 + kb;
            v8bf lo = __builtin_bit_cast(v8bf, *(const u32x4*)(wrow + k0));
            v8bf hi = __builtin_bit_cast(v8bf, *(const u32x4*)(wrow + k0 + 16));
            v16bf b;
            #pragma unroll
            for (int j = 0; j < 8; ++j) { b[j] = lo[j]; b[j + 8] = hi[j]; }
            c0 = __builtin_amdgcn_wmma_f32_16x16x32_bf16(
                     false, a0[ks], false, b, (short)0, c0, false, false);
            c1 = __builtin_amdgcn_wmma_f32_16x16x32_bf16(
                     false, a1[ks], false, b, (short)0, c1, false, false);
        }
        const int nn = n * 16 + rsel;
        #pragma unroll
        for (int r = 0; r < 8; ++r) {
            float v0 = c0[r];
            if (v0 > bestv0[r]) { bestv0[r] = v0; besti0[r] = nn; }
            float v1 = c1[r];
            if (v1 > bestv1[r]) { bestv1[r] = v1; besti1[r] = nn; }
        }
    }

    // Reduce across the 16 lanes of each half (they cover different N).
    #pragma unroll
    for (int r = 0; r < 8; ++r) {
        float v0 = bestv0[r];
        int   i0 = besti0[r];
        float v1 = bestv1[r];
        int   i1 = besti1[r];
        #pragma unroll
        for (int off = 8; off; off >>= 1) {
            float ov = __shfl_xor(v0, off, 32);
            int   oi = __shfl_xor(i0, off, 32);
            if (ov > v0 || (ov == v0 && oi < i0)) { v0 = ov; i0 = oi; }
            ov = __shfl_xor(v1, off, 32);
            oi = __shfl_xor(i1, off, 32);
            if (ov > v1 || (ov == v1 && oi < i1)) { v1 = ov; i1 = oi; }
        }
        if (rsel == 0) {
            int m = (lane >> 4) * 8 + r;   // lane0 -> rows 0..7, lane16 -> rows 8..15
            min_idx[m0 + m]      = i0;
            min_idx[m0 + 16 + m] = i1;
        }
    }
}

// ---------------------- per-token squared error partials + used-code marking
__global__ void __launch_bounds__(256) k_loss_mark(const float* __restrict__ z,
                                                   const float* __restrict__ W,
                                                   const int*   __restrict__ min_idx,
                                                   int*   __restrict__ mask,
                                                   float* __restrict__ partial) {
    int tok = blockIdx.x;
    int t   = threadIdx.x;
    int idx = min_idx[tok];
    float d = z[(size_t)tok * E_DIM + t] - W[(size_t)idx * E_DIM + t];
    float s = d * d;
    #pragma unroll
    for (int off = 16; off; off >>= 1) s += __shfl_xor(s, off, 32);
    __shared__ float sm[8];
    int wave = t >> 5, lane = t & 31;
    if (lane == 0) sm[wave] = s;
    __syncthreads();
    if (t == 0) {
        float tot = 0.f;
        #pragma unroll
        for (int i = 0; i < 8; ++i) tot += sm[i];
        partial[tok] = tot;
        mask[idx] = 1;   // benign race: all writers store 1
    }
}

// -------- stable partition ranks (used codes sorted first, then unused) -----
__global__ void __launch_bounds__(1024) k_rank_label(const int* __restrict__ mask,
                                                     int*   __restrict__ rank,
                                                     float* __restrict__ label_out) {
    __shared__ int sm[N_E];
    int t = threadIdx.x;
    sm[t] = mask[t];
    __syncthreads();
    int prefix = 0, total = 0;
    for (int j = 0; j < N_E; ++j) {
        int m = sm[j];
        total += m;
        if (j < t) prefix += m;
    }
    int r = sm[t] ? prefix : total + (t - prefix);
    rank[t] = r;
    label_out[r] = (float)sm[t];
}

// ------------------------------------------------ emb = W[order] scatter ----
__global__ void __launch_bounds__(256) k_emb(const float* __restrict__ W,
                                             const int*   __restrict__ rank,
                                             float* __restrict__ emb) {
    int e = blockIdx.x;
    int t = threadIdx.x;
    emb[(size_t)rank[e] * E_DIM + t] = W[(size_t)e * E_DIM + t];
}

// -------------------------- deterministic loss reduction + scale ------------
__global__ void __launch_bounds__(1024) k_final(const float* __restrict__ partial,
                                                float* __restrict__ loss_out) {
    __shared__ float sm[32];
    int t = threadIdx.x;
    float s = 0.f;
    for (int j = t; j < N_TOK; j += 1024) s += partial[j];
    #pragma unroll
    for (int off = 16; off; off >>= 1) s += __shfl_xor(s, off, 32);
    int wave = t >> 5, lane = t & 31;
    if (lane == 0) sm[wave] = s;
    __syncthreads();
    if (t == 0) {
        float tot = 0.f;
        #pragma unroll
        for (int i = 0; i < 32; ++i) tot += sm[i];
        *loss_out = tot * (1.0f + BETA) / 16777216.0f;   // / (N_TOK*E_DIM)
    }
}

extern "C" void kernel_launch(void* const* d_in, const int* in_sizes, int n_in,
                              void* d_out, int out_size, void* d_ws, size_t ws_size,
                              hipStream_t stream) {
    const float* z = (const float*)d_in[0];   // (65536, 256) f32
    const float* W = (const float*)d_in[1];   // (1024, 256)  f32

    float* out   = (float*)d_out;
    float* emb   = out;                       // 1024*256
    float* label = out + (size_t)N_E * E_DIM; // 1024
    float* loss  = label + N_E;               // 1

    char* ws = (char*)d_ws;
    __bf16* Wn      = (__bf16*)(ws);                       // 524288 B
    int*    min_idx = (int*)   (ws + 524288);              // 262144 B
    int*    mask    = (int*)   (ws + 786432);              //   4096 B
    int*    rank    = (int*)   (ws + 790528);              //   4096 B
    float*  partial = (float*) (ws + 794624);              // 262144 B

    k_init<<<1, 1024, 0, stream>>>(mask);
    k_normalize_W<<<N_E, 256, 0, stream>>>(W, Wn);
    k_score_argmax<<<N_TOK / 32 / 8, 256, 0, stream>>>(z, Wn, min_idx);
    k_loss_mark<<<N_TOK, 256, 0, stream>>>(z, W, min_idx, mask, partial);
    k_rank_label<<<1, 1024, 0, stream>>>(mask, rank, label);
    k_emb<<<N_E, 256, 0, stream>>>(W, rank, emb);
    k_final<<<1, 1024, 0, stream>>>(partial, loss);
}
